// QResnetBlock_20194936225951
// MI455X (gfx1250) — matrix operand order repaired and checked by reading, not compiled, not run on hardware
//
#include <hip/hip_runtime.h>
#include <hip/hip_bf16.h>
#include <stdint.h>

// ---------------------------------------------------------------------------
// QResnetBlock for MI455X (gfx1250, wave32, WMMA).
// Convs -> implicit GEMM with v_wmma_f32_16x16x32_bf16 (f32 accumulate).
// LDS staging via GLOBAL_LOAD_ASYNC_TO_LDS_B128 (ASYNCcnt / s_wait_asynccnt).
// ---------------------------------------------------------------------------

typedef __attribute__((ext_vector_type(16))) __bf16 v16bf;
typedef __attribute__((ext_vector_type(8)))  float  v8f;

union FragAB { v16bf v; unsigned int u[8]; };
union FragC  { v8f  v; float f[8]; };

#define B_   16
#define H_   64
#define W_   64
#define C_   128
#define QC_  12
#define CC_  116   // C_ - QC_
#define T_   512

__device__ __forceinline__ float sigmoid_f(float x) { return 1.f / (1.f + __expf(-x)); }

__device__ __forceinline__ unsigned int pack_bf16x2(float a, float b) {
  unsigned int ua = __float_as_uint(a); ua += 0x7FFFu + ((ua >> 16) & 1u);
  unsigned int ub = __float_as_uint(b); ub += 0x7FFFu + ((ub >> 16) & 1u);
  return (ua >> 16) | (ub & 0xFFFF0000u);
}

// Generic LDS pointer -> 32-bit LDS byte offset (low 32 bits of the aperture
// address are the intra-allocation LDS address on gfx1250).
__device__ __forceinline__ unsigned lds_off_u32(const void* p) {
  return (unsigned)(uintptr_t)p;
}

// Async DMA: 16 bytes per active lane, global (SADDR + VADDR) -> LDS[ldsOff].
// Tracked by ASYNCcnt; drained with s_wait_asynccnt.
__device__ __forceinline__ void async_ld_b128(unsigned ldsOff, unsigned gOff, const void* base) {
  asm volatile("global_load_async_to_lds_b128 %0, %1, %2"
               :
               : "v"(ldsOff), "v"(gOff), "s"(base)
               : "memory");
}
__device__ __forceinline__ void wait_async0() {
  asm volatile("s_wait_asynccnt 0x0" ::: "memory");
}

// ---------------------------------------------------------------------------
// 4-qubit circuit: RX data layer, RX theta layer, CNOT ring, <Z_w> readout.
// State index k = w0*8 + w1*4 + w2*2 + w3 (wire w at bit 3-w).
// ---------------------------------------------------------------------------
__device__ void apply_rx(float* re, float* im, float phi, int w) {
  float c = cosf(0.5f * phi);
  float s = sinf(0.5f * phi);
  int stride = 1 << (3 - w);
  for (int k = 0; k < 16; ++k) {
    if (k & stride) continue;
    int k1 = k | stride;
    float r0 = re[k], i0 = im[k], r1 = re[k1], i1 = im[k1];
    re[k]  =  c * r0 + s * i1;
    im[k]  =  c * i0 - s * r1;
    re[k1] =  s * i0 + c * r1;
    im[k1] = -s * r0 + c * i1;
  }
}

__device__ void apply_cnot(float* re, float* im, int c, int t) {
  int mc = 1 << (3 - c), mt = 1 << (3 - t);
  for (int k = 0; k < 16; ++k) {
    if (!(k & mc)) continue;
    if (k & mt) continue;
    int k1 = k | mt;
    float tr = re[k]; re[k] = re[k1]; re[k1] = tr;
    float ti = im[k]; im[k] = im[k1]; im[k1] = ti;
  }
}

__device__ void run_circuit(const float x4[4], const float th[4], float z[4]) {
  float re[16], im[16];
  for (int k = 0; k < 16; ++k) { re[k] = 0.f; im[k] = 0.f; }
  re[0] = 1.f;
  for (int w = 0; w < 4; ++w) apply_rx(re, im, x4[w], w);
  for (int w = 0; w < 4; ++w) apply_rx(re, im, th[w], w);
  apply_cnot(re, im, 0, 1);
  apply_cnot(re, im, 1, 2);
  apply_cnot(re, im, 2, 3);
  apply_cnot(re, im, 3, 0);
  for (int w = 0; w < 4; ++w) {
    float acc = 0.f;
    for (int k = 0; k < 16; ++k) {
      float p = re[k] * re[k] + im[k] * im[k];
      acc += ((k >> (3 - w)) & 1) ? -p : p;
    }
    z[w] = acc;
  }
}

// ---------------------------------------------------------------------------
// K1: pooled[b,c] = mean_{H,W} x[b,:,:,c]   (c < 12). One block per (b,c).
// ---------------------------------------------------------------------------
__global__ void pooled_kernel(const float* __restrict__ x, float* __restrict__ pooled) {
  __shared__ float s[256];
  int bc = blockIdx.x;
  int b = bc / QC_, c = bc % QC_;
  const float* xp = x + (size_t)b * H_ * W_ * C_ + c;
  float acc = 0.f;
  for (int p = threadIdx.x; p < H_ * W_; p += 256) acc += xp[(size_t)p * C_];
  s[threadIdx.x] = acc; __syncthreads();
  for (int st = 128; st > 0; st >>= 1) {
    if (threadIdx.x < st) s[threadIdx.x] += s[threadIdx.x + st];
    __syncthreads();
  }
  if (threadIdx.x == 0) pooled[bc] = s[0] * (1.f / (H_ * W_));
}

// ---------------------------------------------------------------------------
// K2: t[b,:] = swish(emb[b,:]) @ time_w + time_b   (16 x 512 x 256)
// ---------------------------------------------------------------------------
__global__ void timemlp_kernel(const float* __restrict__ emb, const float* __restrict__ w,
                               const float* __restrict__ bias, float* __restrict__ tvec) {
  int b = blockIdx.x, j = threadIdx.x;
  const float* e = emb + (size_t)b * T_;
  float acc = bias[j];
  for (int k = 0; k < T_; ++k) {
    float v = e[k];
    acc += (v * sigmoid_f(v)) * w[(size_t)k * 256 + j];
  }
  tvec[b * 256 + j] = acc;
}

// ---------------------------------------------------------------------------
// K3: quantum block #1  (48 independent circuits)
// ---------------------------------------------------------------------------
__global__ void quantum1_kernel(const float* __restrict__ pooled, const float* __restrict__ qp,
                                float* __restrict__ qout) {
  int i = threadIdx.x;
  if (i >= B_ * 3) return;
  int b = i / 3, g = i % 3;
  float x4[4], th[4], z[4];
  for (int w = 0; w < 4; ++w) { x4[w] = pooled[b * QC_ + 4 * g + w]; th[w] = qp[g * 4 + w]; }
  run_circuit(x4, th, z);
  for (int w = 0; w < 4; ++w) qout[b * QC_ + 4 * g + w] = z[w];
}

// ---------------------------------------------------------------------------
// K4: weight-standardize + bf16 pack: (3,3,116,116) -> [9][co:128][ci pairs:64]
// One block per padded output channel co (128 blocks).
// ---------------------------------------------------------------------------
__global__ void wstd_pack_kernel(const float* __restrict__ K, unsigned int* __restrict__ wp) {
  __shared__ float s1[256], s2[256];
  int co = blockIdx.x;
  float a1 = 0.f, a2 = 0.f;
  if (co < CC_) {
    for (int i = threadIdx.x; i < 9 * CC_; i += 256) {
      float v = K[(size_t)i * CC_ + co];
      a1 += v; a2 += v * v;
    }
  }
  s1[threadIdx.x] = a1; s2[threadIdx.x] = a2; __syncthreads();
  for (int st = 128; st > 0; st >>= 1) {
    if (threadIdx.x < st) { s1[threadIdx.x] += s1[threadIdx.x + st]; s2[threadIdx.x] += s2[threadIdx.x + st]; }
    __syncthreads();
  }
  float mean = s1[0] * (1.f / (9 * CC_));
  float var  = s2[0] * (1.f / (9 * CC_)) - mean * mean;
  float rinv = rsqrtf(var + 1e-5f);
  for (int j = threadIdx.x; j < 9 * 64; j += 256) {
    int tap = j >> 6, cp = j & 63, ci0 = 2 * cp;
    float v0 = 0.f, v1 = 0.f;
    if (co < CC_) {
      if (ci0     < CC_) v0 = (K[((size_t)tap * CC_ + ci0    ) * CC_ + co] - mean) * rinv;
      if (ci0 + 1 < CC_) v1 = (K[((size_t)tap * CC_ + ci0 + 1) * CC_ + co] - mean) * rinv;
    }
    wp[((size_t)tap * 128 + co) * 64 + cp] = pack_bf16x2(v0, v1);
  }
}

// ---------------------------------------------------------------------------
// K5: pack x[...,12:] -> padded bf16 image (B,66,66,64 uints). Zero halo/tail.
// One block per (b, padded row).
// ---------------------------------------------------------------------------
__global__ void apack_x_kernel(const float* __restrict__ x, unsigned int* __restrict__ A) {
  int blk = blockIdx.x;
  int b = blk / 66, pr = blk % 66;
  unsigned int* dst = A + ((size_t)b * 66 + pr) * 66 * 64;
  int h = pr - 1;
  for (int i = threadIdx.x; i < 66 * 64; i += 256) {
    int pc = i >> 6, cp = i & 63, w = pc - 1;
    unsigned int val = 0u;
    if (h >= 0 && h < H_ && w >= 0 && w < W_) {
      int a0 = 2 * cp;
      const float* xp = x + ((size_t)(b * H_ + h) * W_ + w) * C_;
      float v0 = (a0     < CC_) ? xp[QC_ + a0    ] : 0.f;
      float v1 = (a0 + 1 < CC_) ? xp[QC_ + a0 + 1] : 0.f;
      val = pack_bf16x2(v0, v1);
    }
    dst[i] = val;
  }
}

// ---------------------------------------------------------------------------
// K6/K10: implicit-GEMM 3x3 conv via WMMA bf16.
// Block = one (b, output row): M=64 pixels, N=128 (116 real co), K=9*128.
// 8 waves: 4 M-tiles x 2 N-halves; each wave: 4 accumulators 16x16.
// LDS staged with GLOBAL_LOAD_ASYNC_TO_LDS_B128.
// Cell stride 68 uints = 272 B: 16B-aligned chunks AND bank-conflict-free
// (68 mod 64 banks = 4 -> 16 consecutive cells hit 16 distinct banks).
// ---------------------------------------------------------------------------
#define ACELLS  (3 * 66)
#define CSTRIDE 68
__global__ __launch_bounds__(256) void conv_wmma_kernel(
    const unsigned int* __restrict__ A,   // (B,66,66,64) packed bf16x2
    const unsigned int* __restrict__ Wp,  // (9,128,64)   packed bf16x2 [tap][co][ci]
    const float* __restrict__ bias,       // (116)
    float* __restrict__ Out)              // (B,64,64,116) f32, bias included
{
  __shared__ __align__(16) unsigned int aLds[ACELLS * CSTRIDE];
  __shared__ __align__(16) unsigned int bLds[128 * CSTRIDE];

  const int bh   = blockIdx.x;
  const int b    = bh >> 6;
  const int h    = bh & 63;
  const int tid  = threadIdx.x;
  const int lane = tid & 31;
  const int wv   = tid >> 5;
  const int mOff   = (wv & 3) * 16;
  const int nHalf  = (wv >> 2) * 64;
  const int row16  = lane & 15;
  const int hiHalf = lane >> 4;

  const unsigned aLdsBase = lds_off_u32(&aLds[0]);
  const unsigned bLdsBase = lds_off_u32(&bLds[0]);

  // stage 3 padded input rows h..h+2 (async DMA, 16B chunks; cell = 64 uints)
  const void* aRow = (const void*)(A + ((size_t)b * 66 + h) * 66 * 64);
  for (int ch = tid; ch < (ACELLS * 64) / 4; ch += 256) {          // 3168 chunks
    int cell = ch >> 4, wi = ch & 15;
    async_ld_b128(aLdsBase + (unsigned)(cell * (CSTRIDE * 4) + wi * 16),
                  (unsigned)(ch * 16), aRow);
  }

  FragC acc[4];
#pragma unroll
  for (int t = 0; t < 4; ++t)
#pragma unroll
    for (int j = 0; j < 8; ++j) acc[t].f[j] = 0.f;

  for (int tap = 0; tap < 9; ++tap) {
    __syncthreads();                       // previous tap's compute done
    const void* wSrc = (const void*)(Wp + (size_t)tap * 128 * 64);
    for (int ch = tid; ch < (128 * 64) / 4; ch += 256) {           // 2048 chunks
      int cell = ch >> 4, wi = ch & 15;
      async_ld_b128(bLdsBase + (unsigned)(cell * (CSTRIDE * 4) + wi * 16),
                    (unsigned)(ch * 16), wSrc);
    }
    if (tap < 8) __builtin_prefetch(Wp + (size_t)(tap + 1) * 128 * 64 + tid * 16, 0, 1);
    wait_async0();                          // drain this wave's async DMAs
    __syncthreads();                        // all waves' tiles visible

    const int kh = tap / 3, kw = tap - 3 * kh;
    const int m = mOff + row16;                       // this lane's A row (pixel)
    const unsigned int* apBase = &aLds[(kh * 66 + m + kw) * CSTRIDE];

#pragma unroll
    for (int kc = 0; kc < 4; ++kc) {                  // 32 channels per WMMA
      // A fragment, ISA 16-bit 16x32 layout:
      //  lanes<16: K pairs {0..7}+{16..23}; lanes>=16: {8..15}+{24..31}
      FragAB afr;
      const unsigned int* ap = apBase + kc * 16 + hiHalf * 4;
#pragma unroll
      for (int e = 0; e < 4; ++e) afr.u[e]     = ap[e];
#pragma unroll
      for (int e = 0; e < 4; ++e) afr.u[4 + e] = ap[8 + e];

#pragma unroll
      for (int t = 0; t < 4; ++t) {
        // B fragment (32x16): col = nHalf+t*16+row16; K = lanes<16 ? {0..7}+{16..23} : {8..15}+{24..31}
        const int co = nHalf + t * 16 + row16;
        const unsigned int* bp = &bLds[co * CSTRIDE + kc * 16 + hiHalf * 4];
        FragAB bfr;
#pragma unroll
        for (int e = 0; e < 4; ++e) { bfr.u[e] = bp[e]; bfr.u[4 + e] = bp[8 + e]; }
        acc[t].v = __builtin_amdgcn_wmma_f32_16x16x32_bf16(
            false, afr.v, false, bfr.v, (short)0, acc[t].v, false, false);
      }
    }
  }

  // epilogue: C/D layout: VGPR j -> M = j + 8*hiHalf, N = lane%16
#pragma unroll
  for (int t = 0; t < 4; ++t) {
    const int n = nHalf + t * 16 + row16;
    if (n < CC_) {
      const float bv = bias[n];
      float* op = Out + ((size_t)(b * H_ + h) * W_) * CC_ + n;
#pragma unroll
      for (int j = 0; j < 8; ++j) {
        const int m2 = mOff + j + hiHalf * 8;
        op[(size_t)m2 * CC_] = acc[t].f[j] + bv;
      }
    }
  }
}

// ---------------------------------------------------------------------------
// K7/K11: group-norm stats per (b,g): mean/rstd over H*W*16 of
// concat(qvec broadcast (c<12), conv (c>=12)).
// ---------------------------------------------------------------------------
__global__ void gnstats_kernel(const float* __restrict__ conv, const float* __restrict__ qv,
                               float* __restrict__ mu, float* __restrict__ rstd) {
  __shared__ float s1[256], s2[256];
  int blk = blockIdx.x;
  int b = blk >> 3, g = blk & 7;
  float a1 = 0.f, a2 = 0.f;
  for (int i = threadIdx.x; i < H_ * W_ * 16; i += 256) {
    int cl = i & 15, pix = i >> 4;
    int c = g * 16 + cl;
    float v = (c < QC_) ? qv[b * QC_ + c]
                        : conv[((size_t)b * H_ * W_ + pix) * CC_ + (c - QC_)];
    a1 += v; a2 += v * v;
  }
  s1[threadIdx.x] = a1; s2[threadIdx.x] = a2; __syncthreads();
  for (int st = 128; st > 0; st >>= 1) {
    if (threadIdx.x < st) { s1[threadIdx.x] += s1[threadIdx.x + st]; s2[threadIdx.x] += s2[threadIdx.x + st]; }
    __syncthreads();
  }
  if (threadIdx.x == 0) {
    float m = s1[0] * (1.f / 65536.f);
    float v = s2[0] * (1.f / 65536.f) - m * m;
    mu[blk] = m;
    rstd[blk] = rsqrtf(v + 1e-6f);
  }
}

// ---------------------------------------------------------------------------
// K8: fused GN0 + FiLM + swish for channels 12..127, re-pack as padded bf16
// image for conv1. One block per (b, padded row).
// ---------------------------------------------------------------------------
__global__ void gn0_apply_pack_kernel(const float* __restrict__ conv, const float* __restrict__ mu,
                                      const float* __restrict__ rstd, const float* __restrict__ gns,
                                      const float* __restrict__ gnb, const float* __restrict__ tvec,
                                      unsigned int* __restrict__ A) {
  int blk = blockIdx.x;
  int b = blk / 66, pr = blk % 66;
  unsigned int* dst = A + ((size_t)b * 66 + pr) * 66 * 64;
  int h = pr - 1;
  for (int i = threadIdx.x; i < 66 * 64; i += 256) {
    int pc = i >> 6, cp = i & 63, w = pc - 1;
    unsigned int val = 0u;
    if (h >= 0 && h < H_ && w >= 0 && w < W_) {
      float v2[2] = {0.f, 0.f};
#pragma unroll
      for (int q = 0; q < 2; ++q) {
        int a = 2 * cp + q;
        if (a < CC_) {
          int c = QC_ + a;
          int g = c >> 4;
          float raw = conv[((size_t)(b * H_ + h) * W_ + w) * CC_ + a];
          float hn = (raw - mu[b * 8 + g]) * rstd[b * 8 + g] * gns[c] + gnb[c];
          float vv = hn * (1.f + tvec[b * 256 + c]) + tvec[b * 256 + 128 + c];
          v2[q] = vv * sigmoid_f(vv);
        }
      }
      val = pack_bf16x2(v2[0], v2[1]);
    }
    dst[i] = val;
  }
}

// ---------------------------------------------------------------------------
// K9: quantum block #2. h[...,:12] is spatially constant, so its pooled mean
// is computed analytically from qout + GN0 stats + FiLM — no big reduction.
// ---------------------------------------------------------------------------
__global__ void quantum2_kernel(const float* __restrict__ qout, const float* __restrict__ qp,
                                const float* __restrict__ mu0, const float* __restrict__ rstd0,
                                const float* __restrict__ gns, const float* __restrict__ gnb,
                                const float* __restrict__ tvec, float* __restrict__ qout2) {
  int i = threadIdx.x;
  if (i >= B_ * 3) return;
  int b = i / 3, g = i % 3;
  float m = mu0[b * 8 + 0], r = rstd0[b * 8 + 0];   // c<12 -> group 0
  float x4[4], th[4], z[4];
  for (int w = 0; w < 4; ++w) {
    int c = 4 * g + w;
    float hn = (qout[b * QC_ + c] - m) * r * gns[c] + gnb[c];
    float v = hn * (1.f + tvec[b * 256 + c]) + tvec[b * 256 + 128 + c];
    x4[w] = v * sigmoid_f(v);
    th[w] = qp[g * 4 + w];
  }
  run_circuit(x4, th, z);
  for (int w = 0; w < 4; ++w) qout2[b * QC_ + 4 * g + w] = z[w];
}

// ---------------------------------------------------------------------------
// K12: out = x + swish(GN1(concat(qout2 bcast, conv1_out)))
// ---------------------------------------------------------------------------
__global__ void final_kernel(const float* __restrict__ x, const float* __restrict__ conv,
                             const float* __restrict__ qv, const float* __restrict__ mu,
                             const float* __restrict__ rstd, const float* __restrict__ gns,
                             const float* __restrict__ gnb, float* __restrict__ out) {
  size_t idx = (size_t)blockIdx.x * 256 + threadIdx.x;
  int c = (int)(idx & 127);
  size_t pix = idx >> 7;
  int b = (int)(pix >> 12);
  int p = (int)(pix & 4095);
  float pre = (c < QC_) ? qv[b * QC_ + c]
                        : conv[((size_t)b * 4096 + p) * CC_ + (c - QC_)];
  int g = c >> 4;
  float hn = (pre - mu[b * 8 + g]) * rstd[b * 8 + g] * gns[c] + gnb[c];
  out[idx] = x[idx] + hn * sigmoid_f(hn);
}

// ---------------------------------------------------------------------------
extern "C" void kernel_launch(void* const* d_in, const int* in_sizes, int n_in,
                              void* d_out, int out_size, void* d_ws, size_t ws_size,
                              hipStream_t stream) {
  (void)in_sizes; (void)n_in; (void)out_size; (void)ws_size;

  const float* x        = (const float*)d_in[0];
  const float* time_emb = (const float*)d_in[1];
  const float* qparams  = (const float*)d_in[2];
  const float* k0       = (const float*)d_in[3];
  const float* b0       = (const float*)d_in[4];
  const float* k1       = (const float*)d_in[5];
  const float* b1       = (const float*)d_in[6];
  const float* gn0s     = (const float*)d_in[7];
  const float* gn0b     = (const float*)d_in[8];
  const float* gn1s     = (const float*)d_in[9];
  const float* gn1b     = (const float*)d_in[10];
  const float* tw       = (const float*)d_in[11];
  const float* tb       = (const float*)d_in[12];
  float* out = (float*)d_out;

  char* ws = (char*)d_ws;
  size_t off = 0;
  auto take = [&](size_t bytes) -> char* {
    char* p = ws + off;
    off = (off + bytes + 255) & ~(size_t)255;
    return p;
  };

  // Big buffers are reused across the two convs (stream serializes kernels).
  unsigned int* Abuf    = (unsigned int*)take((size_t)B_ * 66 * 66 * 64 * 4); // 17.8 MB
  float*        convout = (float*)take((size_t)B_ * H_ * W_ * CC_ * 4);       // 30.4 MB
  unsigned int* wp0     = (unsigned int*)take((size_t)9 * 128 * 64 * 4);
  unsigned int* wp1     = (unsigned int*)take((size_t)9 * 128 * 64 * 4);
  float* tvec   = (float*)take((size_t)B_ * 256 * 4);
  float* pooled = (float*)take((size_t)B_ * QC_ * 4);
  float* qout   = (float*)take((size_t)B_ * QC_ * 4);
  float* qout2  = (float*)take((size_t)B_ * QC_ * 4);
  float* mu0    = (float*)take((size_t)B_ * 8 * 4);
  float* rstd0  = (float*)take((size_t)B_ * 8 * 4);
  float* mu1    = (float*)take((size_t)B_ * 8 * 4);
  float* rstd1  = (float*)take((size_t)B_ * 8 * 4);

  // Small / prep kernels
  pooled_kernel<<<B_ * QC_, 256, 0, stream>>>(x, pooled);
  timemlp_kernel<<<B_, 256, 0, stream>>>(time_emb, tw, tb, tvec);
  quantum1_kernel<<<1, 64, 0, stream>>>(pooled, qparams, qout);
  wstd_pack_kernel<<<128, 256, 0, stream>>>(k0, wp0);
  wstd_pack_kernel<<<128, 256, 0, stream>>>(k1, wp1);
  apack_x_kernel<<<B_ * 66, 256, 0, stream>>>(x, Abuf);

  // Conv0 (WMMA) + GN0
  conv_wmma_kernel<<<B_ * H_, 256, 0, stream>>>(Abuf, wp0, b0, convout);
  gnstats_kernel<<<B_ * 8, 256, 0, stream>>>(convout, qout, mu0, rstd0);
  gn0_apply_pack_kernel<<<B_ * 66, 256, 0, stream>>>(convout, mu0, rstd0, gn0s, gn0b, tvec, Abuf);
  quantum2_kernel<<<1, 64, 0, stream>>>(qout, qparams, mu0, rstd0, gn0s, gn0b, tvec, qout2);

  // Conv1 (WMMA) + GN1 + residual
  conv_wmma_kernel<<<B_ * H_, 256, 0, stream>>>(Abuf, wp1, b1, convout);
  gnstats_kernel<<<B_ * 8, 256, 0, stream>>>(convout, qout2, mu1, rstd1);
  final_kernel<<<(B_ * H_ * W_ * C_) / 256, 256, 0, stream>>>(
      x, convout, qout2, mu1, rstd1, gn1s, gn1b, out);
}
